// LSTM_76759655514333
// MI455X (gfx1250) — matrix operand order repaired and checked by reading, not compiled, not run on hardware
//
#include <hip/hip_runtime.h>
#include <stdint.h>

// ---------------------------------------------------------------------------
// LSTM on MI455X (gfx1250): bf16 WMMA GEMMs, TDM weight preload,
// async global->LDS h-exchange, cluster barrier + atomic fallback sync.
// ---------------------------------------------------------------------------

typedef __attribute__((ext_vector_type(16))) __bf16 v16bf;
typedef __attribute__((ext_vector_type(8)))  float  v8f;
typedef __attribute__((ext_vector_type(4)))  unsigned int v4u;
typedef __attribute__((ext_vector_type(8)))  int v8i;
typedef __attribute__((ext_vector_type(4)))  int v4i;

union Frag16 { v16bf v; uint4 q[2]; };

__device__ __forceinline__ uint16_t f2bf(float f) {
  uint32_t u = __float_as_uint(f);
  u += 0x7FFFu + ((u >> 16) & 1u);          // round-to-nearest-even
  return (uint16_t)(u >> 16);
}
__device__ __forceinline__ float bf2f(uint16_t h) {
  return __uint_as_float(((uint32_t)h) << 16);
}
__device__ __forceinline__ uint32_t lds_off(const void* p) {
  // generic LDS pointer = {SHARED_BASE_hi32, offset32}; low 32 bits = offset
  return (uint32_t)(uintptr_t)p;
}
// CDNA5 async copy: global -> LDS, 16B per lane, tracked by ASYNCcnt
__device__ __forceinline__ void async_g2l_b128(void* lds, const void* g) {
  asm volatile("global_load_async_to_lds_b128 %0, %1, off"
               :: "v"(lds_off(lds)), "v"(g) : "memory");
}
__device__ __forceinline__ void wait_async0() {
  asm volatile("s_wait_asynccnt 0" ::: "memory");
}
__device__ __forceinline__ float sigm(float x) { return 1.f / (1.f + __expf(-x)); }
__device__ __forceinline__ float tanh_f(float x) {
  float e = __expf(2.f * x);
  return 1.f - 2.f / (e + 1.f);
}

// ---------------------------------------------------------------------------
// small utility kernels
// ---------------------------------------------------------------------------
__global__ void k_zero_i32(int* p, int n) {
  int i = blockIdx.x * blockDim.x + threadIdx.x;
  if (i < n) p[i] = 0;
}
__global__ void k_cvt_bf16(const float* __restrict__ in, uint16_t* __restrict__ out, int n) {
  int i = blockIdx.x * blockDim.x + threadIdx.x;
  int s = gridDim.x * blockDim.x;
  for (; i < n; i += s) out[i] = f2bf(in[i]);
}
__global__ void k_addv(const float* a, const float* b, float* o, int n) {
  int i = blockIdx.x * blockDim.x + threadIdx.x;
  if (i < n) o[i] = a[i] + b[i];
}

// ---------------------------------------------------------------------------
// xg = Xb[M][K] @ Wb[2048][K]^T + bias[n]   (bf16 inputs, fp32 out)
// 128x128 tile / WG, 8 waves, each wave 64x32 (4x2 WMMA tiles), K chunks of 32
// ---------------------------------------------------------------------------
__global__ __launch_bounds__(256) void k_xg_gemm(
    const uint16_t* __restrict__ Xb, const uint16_t* __restrict__ Wb,
    const float* __restrict__ bias, float* __restrict__ xg, int K)
{
  extern __shared__ char smem[];
  uint16_t* At = (uint16_t*)smem;           // [128][32] bf16
  uint16_t* Bt = (uint16_t*)(smem + 8192);  // [128][32] bf16

  const int tid = threadIdx.x;
  const int w = tid >> 5, l = tid & 31;
  const int lm = l & 15, kh = l >> 4;
  const int m0 = blockIdx.y * 128;
  const int n0 = blockIdx.x * 128;
  const int wm = (w & 1) * 64;
  const int wn = (w >> 1) * 32;

  v8f acc[4][2];
  for (int i = 0; i < 4; ++i)
    for (int j = 0; j < 2; ++j)
      for (int r = 0; r < 8; ++r) acc[i][j][r] = 0.f;

  const int row = tid >> 1, half = tid & 1;   // tile staging: 32B per thread/tile
  for (int k0 = 0; k0 < K; k0 += 32) {
    const uint16_t* xs = Xb + (size_t)(m0 + row) * K + k0 + half * 16;
    const uint16_t* ws = Wb + (size_t)(n0 + row) * K + k0 + half * 16;
    uint16_t* ad = &At[row * 32 + half * 16];
    uint16_t* bd = &Bt[row * 32 + half * 16];
    async_g2l_b128(ad, xs);
    async_g2l_b128(ad + 8, xs + 8);
    async_g2l_b128(bd, ws);
    async_g2l_b128(bd + 8, ws + 8);
    wait_async0();
    __syncthreads();

    Frag16 bf[2];
    for (int j = 0; j < 2; ++j) {
      const uint16_t* bp = &Bt[(wn + j * 16 + lm) * 32 + 8 * kh];
      bf[j].q[0] = *(const uint4*)bp;
      bf[j].q[1] = *(const uint4*)(bp + 16);
    }
    for (int i = 0; i < 4; ++i) {
      Frag16 af;
      const uint16_t* ap = &At[(wm + i * 16 + lm) * 32];
      af.q[0] = *(const uint4*)(ap + kh * 8);
      af.q[1] = *(const uint4*)(ap + 16 + kh * 8);
      for (int j = 0; j < 2; ++j)
        acc[i][j] = __builtin_amdgcn_wmma_f32_16x16x32_bf16(
            false, af.v, false, bf[j].v, (short)0, acc[i][j], false, false);
    }
    __syncthreads();
  }
  for (int i = 0; i < 4; ++i)
    for (int j = 0; j < 2; ++j) {
      int n = n0 + wn + j * 16 + lm;
      float bs = bias[n];
      for (int r = 0; r < 8; ++r) {
        int m = m0 + wm + i * 16 + r + 8 * kh;
        xg[(size_t)m * 2048 + n] = acc[i][j][r] + bs;
      }
    }
}

// ---------------------------------------------------------------------------
// Recurrence: 4 batch-groups x 16 WGs. Each WG owns 32 hidden units
// (=128 gate cols), W_hh slice LDS-resident via TDM. Per step: WMMA gate GEMM,
// LSTM cell, h exchange through global ping-pong + group barrier.
// ---------------------------------------------------------------------------
__global__ __launch_bounds__(256) void k_lstm_recur(
    const float* __restrict__ xg,       // [64][512][2048] fp32
    const uint16_t* __restrict__ Whh,   // [2048][512] bf16
    const float* __restrict__ h0,       // [64][512]
    const float* __restrict__ c0,       // [64][512]
    uint16_t* __restrict__ outb,        // [64][512][512] bf16
    uint16_t* __restrict__ h_ex,        // [4 groups][2 parity][16][512] bf16
    int* __restrict__ counters)         // [4 groups][512 steps]
{
  extern __shared__ char smem[];
  uint16_t* w_lds = (uint16_t*)smem;                      // [128][512] 128KB
  uint16_t* h_lds = (uint16_t*)(smem + 131072);           // [16][512]  16KB
  float*    gates = (float*)(smem + 131072 + 16384);      // [16][128]   8KB

  const int tid = threadIdx.x;
  const int w = tid >> 5, l = tid & 31;
  const int lm = l & 15, kh = l >> 4;
  const int group = blockIdx.x >> 4;    // batch group of 16 rows
  const int wg    = blockIdx.x & 15;    // 32-hidden-unit slice
  const int bg0 = group * 16;

  // ---- TDM preload of W_hh slice: 3D tile {512 k} x {32 rows} x {4 gates}
  if (w == 0) {
    uint64_t ga = (uint64_t)(uintptr_t)(Whh + (size_t)wg * 32 * 512);
    v4u g0; v8i g1; v4i g2, g3; v8i g4;
    g0[0] = 1u;                                             // count=1
    g0[1] = lds_off(w_lds);                                 // lds_addr
    g0[2] = (uint32_t)ga;                                   // global_addr lo
    g0[3] = (uint32_t)((ga >> 32) & 0x1FFFFFFu) | (2u << 30); // ga hi | type=2
    g1[0] = (1 << 16);                 // data_size=1 (2B)
    g1[1] = (512 << 16);               // tensor_dim0[15:0] @ bits[63:48]
    g1[2] = (2048 << 16);              // tensor_dim1[15:0] @ bits[95:80]
    g1[3] = (512 << 16);               // tile_dim0=512
    g1[4] = 32 | (4 << 16);            // tile_dim1=32, tile_dim2=4
    g1[5] = 512;                       // tensor_dim0_stride
    g1[6] = 0;                         // (262144 & 0xFFFF)<<16 == 0
    g1[7] = 4;                         // tensor_dim1_stride >> 16 (gate stride 512*512)
    g2[0] = 4; g2[1] = 0; g2[2] = 0; g2[3] = 0;   // tensor_dim2=4
    g3[0] = 0; g3[1] = 0; g3[2] = 0; g3[3] = 0;
    for (int i = 0; i < 8; ++i) g4[i] = 0;        // unused extra group (6-arg form)
    __builtin_amdgcn_tensor_load_to_lds(g0, g1, g2, g3, g4, 0);
    __builtin_amdgcn_s_wait_tensorcnt(0);
  }

  // h_lds <- h0 (fp32 -> bf16)
  for (int i = tid; i < 16 * 512; i += 256) {
    int b = i >> 9, k = i & 511;
    h_lds[i] = f2bf(h0[(size_t)(bg0 + b) * 512 + k]);
  }
  // cell state in registers: 16 batch x 32 hidden = 512 vals, 2/thread
  float creg[2];
  for (int r = 0; r < 2; ++r) {
    int p = tid + 256 * r;
    int b = p >> 5, j = p & 31;
    creg[r] = c0[(size_t)(bg0 + b) * 512 + wg * 32 + j];
  }
  __syncthreads();

  for (int t = 0; t < 512; ++t) {
    // ---- gate GEMM slice: wave w -> 16 cols (c = w*16+lm), M=16 batch, K=512
    v8f acc;
    for (int r = 0; r < 8; ++r) acc[r] = 0.f;
    const uint16_t* wrow = &w_lds[(size_t)(w * 16 + lm) * 512];
    const uint16_t* hrow = &h_lds[(size_t)lm * 512];
    for (int kc = 0; kc < 16; ++kc) {
      Frag16 a, b;
      a.q[0] = *(const uint4*)(hrow + kc * 32 + kh * 8);
      a.q[1] = *(const uint4*)(hrow + kc * 32 + 16 + kh * 8);
      b.q[0] = *(const uint4*)(wrow + kc * 32 + kh * 8);
      b.q[1] = *(const uint4*)(wrow + kc * 32 + 16 + kh * 8);
      acc = __builtin_amdgcn_wmma_f32_16x16x32_bf16(
          false, a.v, false, b.v, (short)0, acc, false, false);
    }
    for (int r = 0; r < 8; ++r)
      gates[(r + 8 * kh) * 128 + w * 16 + lm] = acc[r];
    __syncthreads();

    // ---- LSTM cell (fp32): gate col c = gateType*32 + j
    for (int r = 0; r < 2; ++r) {
      int p = tid + 256 * r;
      int b = p >> 5, j = p & 31;
      int jg = wg * 32 + j;
      size_t bt = (size_t)(bg0 + b) * 512 + t;
      const float* xgp = xg + bt * 2048;
      float gi = xgp[jg]        + gates[b * 128 + j];
      float gf = xgp[512 + jg]  + gates[b * 128 + 32 + j];
      float gg = xgp[1024 + jg] + gates[b * 128 + 64 + j];
      float go = xgp[1536 + jg] + gates[b * 128 + 96 + j];
      float iv = sigm(gi), fv = sigm(gf), gv = tanh_f(gg), ov = sigm(go);
      float c = fv * creg[r] + iv * gv;
      creg[r] = c;
      uint16_t hb = f2bf(ov * tanh_f(c));
      outb[bt * 512 + jg] = hb;
      h_ex[((size_t)(group * 2 + (t & 1)) * 16 + b) * 512 + jg] = hb;
    }
    __syncthreads();

    // ---- group-wide barrier over the 16 cooperating WGs
    __builtin_amdgcn_fence(__ATOMIC_RELEASE, "agent");
    if (tid == 0) atomicAdd(&counters[group * 512 + t], 1);
    __builtin_amdgcn_s_cluster_barrier();   // s_barrier_signal/wait -3 (cluster path)
    if (tid == 0) {
      while (__hip_atomic_load(&counters[group * 512 + t],
                               __ATOMIC_RELAXED, __HIP_MEMORY_SCOPE_AGENT) < 16)
        __builtin_amdgcn_s_sleep(1);
    }
    __syncthreads();
    __builtin_amdgcn_fence(__ATOMIC_ACQUIRE, "agent");

    // ---- refresh full h_t into LDS: 16KB async DMA, 64B/thread
    const char* gsrc = (const char*)(h_ex + (size_t)(group * 2 + (t & 1)) * 16 * 512)
                       + tid * 64;
    char* ldst = (char*)h_lds + tid * 64;
    async_g2l_b128(ldst,      gsrc);
    async_g2l_b128(ldst + 16, gsrc + 16);
    async_g2l_b128(ldst + 32, gsrc + 32);
    async_g2l_b128(ldst + 48, gsrc + 48);
    wait_async0();
    __syncthreads();
  }
}

// ---------------------------------------------------------------------------
// y[b,t] = out1[b,t,:] . w_lin + b_lin   (one wave per row)
// ---------------------------------------------------------------------------
__global__ __launch_bounds__(256) void k_proj(
    const uint16_t* __restrict__ out1, const float* __restrict__ w_lin,
    const float* __restrict__ b_lin, float* __restrict__ y)
{
  int wv = threadIdx.x >> 5, l = threadIdx.x & 31;
  int row = blockIdx.x * 8 + wv;
  const uint16_t* p = out1 + (size_t)row * 512;
  float s = 0.f;
  for (int k = l; k < 512; k += 32) s += bf2f(p[k]) * w_lin[k];
  for (int off = 16; off; off >>= 1) s += __shfl_xor(s, off, 32);
  if (l == 0) y[row] = s + b_lin[0];
}

// ---------------------------------------------------------------------------
extern "C" void kernel_launch(void* const* d_in, const int* in_sizes, int n_in,
                              void* d_out, int out_size, void* d_ws, size_t ws_size,
                              hipStream_t stream)
{
  const float* x     = (const float*)d_in[0];
  const float* h0    = (const float*)d_in[1];   // [2][64][512]
  const float* c0    = (const float*)d_in[2];
  const float* w_ih0 = (const float*)d_in[3];
  const float* w_hh0 = (const float*)d_in[4];
  const float* b_ih0 = (const float*)d_in[5];
  const float* b_hh0 = (const float*)d_in[6];
  const float* w_ih1 = (const float*)d_in[7];
  const float* w_hh1 = (const float*)d_in[8];
  const float* b_ih1 = (const float*)d_in[9];
  const float* b_hh1 = (const float*)d_in[10];
  const float* w_lin = (const float*)d_in[11];
  const float* b_lin = (const float*)d_in[12];
  float* y = (float*)d_out;
  (void)in_sizes; (void)n_in; (void)out_size; (void)ws_size;

  char* ws = (char*)d_ws;
  size_t off = 0;
  auto alloc = [&](size_t bytes) {
    char* p = ws + off;
    off += (bytes + 255) & ~(size_t)255;
    return p;
  };
  uint16_t* Xb0   = (uint16_t*)alloc(32768ull * 128 * 2);
  uint16_t* Wih0b = (uint16_t*)alloc(2048ull * 128 * 2);
  uint16_t* Whh0b = (uint16_t*)alloc(2048ull * 512 * 2);
  uint16_t* Wih1b = (uint16_t*)alloc(2048ull * 512 * 2);
  uint16_t* Whh1b = (uint16_t*)alloc(2048ull * 512 * 2);
  float*    bias0 = (float*)alloc(2048 * 4);
  float*    bias1 = (float*)alloc(2048 * 4);
  float*    xgbuf = (float*)alloc(32768ull * 2048 * 4);
  uint16_t* out0  = (uint16_t*)alloc(32768ull * 512 * 2);
  uint16_t* out1  = (uint16_t*)alloc(32768ull * 512 * 2);
  uint16_t* h_ex  = (uint16_t*)alloc(4ull * 2 * 16 * 512 * 2);
  int*      cnts  = (int*)alloc(2ull * 4 * 512 * 4);

  k_zero_i32<<<16, 256, 0, stream>>>(cnts, 4096);
  k_cvt_bf16<<<2048, 256, 0, stream>>>(x,     Xb0,   32768 * 128);
  k_cvt_bf16<<<512,  256, 0, stream>>>(w_ih0, Wih0b, 2048 * 128);
  k_cvt_bf16<<<1024, 256, 0, stream>>>(w_hh0, Whh0b, 2048 * 512);
  k_cvt_bf16<<<1024, 256, 0, stream>>>(w_ih1, Wih1b, 2048 * 512);
  k_cvt_bf16<<<1024, 256, 0, stream>>>(w_hh1, Whh1b, 2048 * 512);
  k_addv<<<8, 256, 0, stream>>>(b_ih0, b_hh0, bias0, 2048);
  k_addv<<<8, 256, 0, stream>>>(b_ih1, b_hh1, bias1, 2048);

  dim3 gg(16, 256);
  const size_t gemm_smem  = 16384;
  const size_t recur_smem = 131072 + 16384 + 8192;

  // Layer 0
  k_xg_gemm<<<gg, 256, gemm_smem, stream>>>(Xb0, Wih0b, bias0, xgbuf, 128);
  k_lstm_recur<<<64, 256, recur_smem, stream>>>(xgbuf, Whh0b, h0, c0,
                                                out0, h_ex, cnts);
  // Layer 1
  k_xg_gemm<<<gg, 256, gemm_smem, stream>>>(out0, Wih1b, bias1, xgbuf, 512);
  k_lstm_recur<<<64, 256, recur_smem, stream>>>(xgbuf, Whh1b, h0 + 32768, c0 + 32768,
                                                out1, h_ex, cnts + 2048);
  // Head
  k_proj<<<4096, 256, 0, stream>>>(out1, w_lin, b_lin, y);
}
